// GCN_63204738728401
// MI455X (gfx1250) — matrix operand order repaired and checked by reading, not compiled, not run on hardware
//
#include <hip/hip_runtime.h>
#include <hip/hip_bf16.h>

#define NNODES 100000
#define NEDGES 1600000
#define DCH 64
#define CCH 40
#define NLAYERS 5

typedef __attribute__((ext_vector_type(16))) __bf16 v16bf;
typedef __attribute__((ext_vector_type(8)))  float  v8f;

// ---------------------------------------------------------------------------
// Convert fp32 weights -> bf16, W_out zero-padded from 64x40 to 64x64.
// Wbf layout: [5][64][64] bf16 row-major.
// ---------------------------------------------------------------------------
__global__ void k_convert_weights(const float* __restrict__ W,
                                  const float* __restrict__ Wout,
                                  __bf16* __restrict__ Wbf) {
  int idx = blockIdx.x * blockDim.x + threadIdx.x;
  if (idx >= NLAYERS * DCH * DCH) return;
  int l  = idx >> 12;          // /4096
  int rc = idx & 4095;
  float v;
  if (l < NLAYERS - 1) {
    v = W[idx];
  } else {
    int r = rc >> 6, c = rc & 63;
    v = (c < CCH) ? Wout[r * CCH + c] : 0.0f;
  }
  Wbf[idx] = (__bf16)v;
}

// ---------------------------------------------------------------------------
// agg[n][c] = bias[c] (c < blen) else 0  -- fuses bias into the scatter init.
// ---------------------------------------------------------------------------
__global__ void k_init_bias(float* __restrict__ agg,
                            const float* __restrict__ bias, int blen) {
  int idx = blockIdx.x * blockDim.x + threadIdx.x;
  if (idx >= NNODES * DCH) return;
  int c = idx & (DCH - 1);
  agg[idx] = (c < blen) ? bias[c] : 0.0f;
}

// ---------------------------------------------------------------------------
// H = (relu?)(X) @ Wbf    X:[N,64] fp32   Wbf:[64,64] bf16   H:[N,64] fp32
// One wave per 16-row tile; 8x v_wmma_f32_16x16x32_bf16 per tile.
// B fragments are loaded once per wave and stay resident (W shared by all).
// RELU is a compile-time flag: no v_cndmask in the inner loop.
// ---------------------------------------------------------------------------
template <bool RELU>
__global__ __launch_bounds__(256)
void k_gemm(const float* __restrict__ X, const __bf16* __restrict__ Wbf,
            float* __restrict__ H) {
  const int lane   = threadIdx.x & 31;
  const int wave   = threadIdx.x >> 5;
  const int gwave  = blockIdx.x * (blockDim.x >> 5) + wave;
  const int nwaves = gridDim.x * (blockDim.x >> 5);

  const int nsel  = lane & 15;   // B/C/D column within tile
  const int khalf = lane >> 4;   // 0 or 1

  // --- B fragments: col tile j in 0..3, K-half kk in 0..1 -----------------
  // B layout (32x16 bf16), matching A: elems 0-7 -> K = khalf*8 + i,
  // elems 8-15 -> K = 16 + khalf*8 + (i-8); column nsel.
  v16bf bfrag[8];
#pragma unroll
  for (int kk = 0; kk < 2; ++kk) {
#pragma unroll
    for (int j = 0; j < 4; ++j) {
      v16bf t;
      const int kb = kk * 32 + khalf * 8;
#pragma unroll
      for (int i = 0; i < 16; ++i)
        t[i] = Wbf[(kb + (i < 8 ? i : i + 8)) * DCH + j * 16 + nsel];
      bfrag[kk * 4 + j] = t;
    }
  }

  // --- A addressing (16x32 bf16 tile): row m = lane&15;
  // lanes 0-15: elems 0-7 -> K+0..7, elems 8-15 -> K+16..23 (kA = 0)
  // lanes 16-31: same with kA = 8.
  const int m  = lane & 15;
  const int kA = khalf * 8;

  for (int tile = gwave; tile < NNODES / 16; tile += nwaves) {
    const float* xrow = X + (size_t)(tile * 16 + m) * DCH;

    v16bf afrag[2];
#pragma unroll
    for (int kk = 0; kk < 2; ++kk) {
      const float* p = xrow + kk * 32 + kA;
      v16bf a;
#pragma unroll
      for (int i = 0; i < 8; ++i) {
        float f = p[i];
        if (RELU) f = fmaxf(f, 0.0f);
        a[i] = (__bf16)f;
      }
#pragma unroll
      for (int i = 0; i < 8; ++i) {
        float f = p[16 + i];
        if (RELU) f = fmaxf(f, 0.0f);
        a[8 + i] = (__bf16)f;
      }
      afrag[kk] = a;
    }

#pragma unroll
    for (int j = 0; j < 4; ++j) {
      v8f acc = {};
      acc = __builtin_amdgcn_wmma_f32_16x16x32_bf16(
          false, afrag[0], false, bfrag[j],     (short)0, acc, false, false);
      acc = __builtin_amdgcn_wmma_f32_16x16x32_bf16(
          false, afrag[1], false, bfrag[4 + j], (short)0, acc, false, false);
      // D layout: lane col = nsel; VGPR r -> row khalf*8 + r.
      float* hp = H + (size_t)(tile * 16 + khalf * 8) * DCH + j * 16 + nsel;
#pragma unroll
      for (int r = 0; r < 8; ++r) hp[r * DCH] = acc[r];
    }
  }
}

// ---------------------------------------------------------------------------
// agg[dst[e]][c] += ew[e] * H[src[e]][c]   (one thread per (edge, channel))
// All 32 lanes of a wave share one edge -> scalarize the edge metadata loads
// via readfirstlane (s_load path); the gather load and the f32 atomic are
// each one fully-dense 128B wave transaction, L2-resident.
// ---------------------------------------------------------------------------
__global__ __launch_bounds__(256)
void k_scatter(const float* __restrict__ H, const int* __restrict__ src,
               const int* __restrict__ dst, const float* __restrict__ ew,
               float* __restrict__ agg) {
  int idx = blockIdx.x * blockDim.x + threadIdx.x;
  if (idx >= NEDGES * DCH) return;
  int e = __builtin_amdgcn_readfirstlane(idx >> 6);  // wave-uniform edge id
  int c = idx & (DCH - 1);
  int s = src[e];
  int d = dst[e];
  float w = ew[e];
  float msg = H[s * DCH + c] * w;
  __hip_atomic_fetch_add(&agg[d * DCH + c], msg,
                         __ATOMIC_RELAXED, __HIP_MEMORY_SCOPE_AGENT);
}

// ---------------------------------------------------------------------------
// out[n][0..39] = agg[n][0..39]  (bias already folded in by k_init_bias)
// ---------------------------------------------------------------------------
__global__ void k_extract(const float* __restrict__ agg, float* __restrict__ out) {
  int idx = blockIdx.x * blockDim.x + threadIdx.x;
  if (idx >= NNODES * CCH) return;
  int n = idx / CCH;
  int c = idx - n * CCH;
  out[idx] = agg[n * DCH + c];
}

// ---------------------------------------------------------------------------
extern "C" void kernel_launch(void* const* d_in, const int* in_sizes, int n_in,
                              void* d_out, int out_size, void* d_ws, size_t ws_size,
                              hipStream_t stream) {
  const float* x    = (const float*)d_in[0];
  const float* ew   = (const float*)d_in[1];
  const float* W    = (const float*)d_in[2];
  const float* b    = (const float*)d_in[3];
  const float* Wout = (const float*)d_in[4];
  const float* bout = (const float*)d_in[5];
  const int*   esrc = (const int*)d_in[6];
  const int*   edst = (const int*)d_in[7];
  float* out = (float*)d_out;

  const size_t nodebuf = (size_t)NNODES * DCH;
  float* h     = (float*)d_ws;
  float* agg_a = h + nodebuf;
  float* agg_b = agg_a + nodebuf;
  __bf16* wbf  = (__bf16*)(agg_b + nodebuf);

  k_convert_weights<<<(NLAYERS * DCH * DCH + 255) / 256, 256, 0, stream>>>(W, Wout, wbf);

  float* aggs[NLAYERS] = {agg_a, agg_b, agg_a, agg_b, agg_a};
  const float* xin = x;

  for (int l = 0; l < NLAYERS; ++l) {
    const __bf16* wl   = wbf + (size_t)l * DCH * DCH;
    const float*  bias = (l < NLAYERS - 1) ? (b + l * DCH) : bout;
    const int     blen = (l < NLAYERS - 1) ? DCH : CCH;
    float* agg = aggs[l];

    k_init_bias<<<(NNODES * DCH + 255) / 256, 256, 0, stream>>>(agg, bias, blen);
    if (l == 0)
      k_gemm<false><<<240, 256, 0, stream>>>(xin, wl, h);
    else
      k_gemm<true><<<240, 256, 0, stream>>>(xin, wl, h);
    k_scatter<<<(NEDGES * DCH + 255) / 256, 256, 0, stream>>>(h, esrc, edst, ew, agg);

    xin = agg;
  }

  k_extract<<<(NNODES * CCH + 255) / 256, 256, 0, stream>>>(aggs[NLAYERS - 1], out);
}
